// BiLSTM_CRF_22608707846775
// MI455X (gfx1250) — compile-verified, matrix operand order checked by reading, not run on hardware
//
#include <hip/hip_runtime.h>
#include <hip/hip_bf16.h>

// ---------------------------------------------------------------------------
// BiLSTM-CRF forward for MI455X (gfx1250, wave32).
//   V=100000 D=256 H=512 H2=256 T=16 L=2048, START=14 STOP=15 NEG=-10000
// Pipeline:
//   1. gather embeds -> f16
//   2. convert Wih_{f,b} and Whh_{f,b} -> f16
//   3. WMMA GEMM: X = embeds @ Wih^T + (bih+bhh)   [2048 x 1024] per direction
//   4. sequential LSTM recurrence, f16 Whh streamed from L2 (mixed-precision
//      FMA, f32 accumulate) + TDM prefetch of next X row into LDS
//   5. feats = [hs_f|hs_b] @ Wt^T + bt             [2048 x 16]
//   6. Viterbi DP (one wave32) + backtrace
// ---------------------------------------------------------------------------

#define L_SEQ 2048
#define D_EMB 256
#define H2    256
#define G4    1024   // 4*H2
#define NTAG  16
#define START_TAG 14
#define STOP_TAG  15
#define NEGV  (-10000.0f)

typedef __attribute__((ext_vector_type(16))) _Float16 v16h;
typedef __attribute__((ext_vector_type(8)))  _Float16 v8h;
typedef __attribute__((ext_vector_type(8)))  float    v8f;
typedef __attribute__((ext_vector_type(4)))  unsigned int v4u;
typedef __attribute__((ext_vector_type(8)))  int      v8i;
typedef __attribute__((ext_vector_type(4)))  int      v4i;

// ---------------- 1. embedding gather + f32->f16 ----------------
__global__ void gather_f16_kernel(const int* __restrict__ sentence,
                                  const float* __restrict__ emb,
                                  _Float16* __restrict__ out) {
    int l = blockIdx.x;          // 0..2047
    int d = threadIdx.x;         // 0..255
    int tok = sentence[l];
    out[l * D_EMB + d] = (_Float16)emb[(long long)tok * D_EMB + d];
}

// ---------------- 2. f32 -> f16 weight convert ----------------
__global__ void cvt_f16_kernel(const float* __restrict__ in,
                               _Float16* __restrict__ out, int n) {
    int i = blockIdx.x * blockDim.x + threadIdx.x;
    if (i < n) out[i] = (_Float16)in[i];
}

// ---------------- 3. WMMA GEMM: X[M=2048,N=1024] = A[M,256] * W[N,256]^T + bias[N]
// One wave per 16x16 output tile; 8 waves per block; K=256 in 8 chunks of 32.
__global__ __launch_bounds__(256)
void gemm_wmma_kernel(const _Float16* __restrict__ A,   // [2048,256] row-major
                      const _Float16* __restrict__ W,   // [1024,256] row-major
                      const float* __restrict__ bih,
                      const float* __restrict__ bhh,
                      float* __restrict__ X) {          // [2048,1024]
    const int lane  = threadIdx.x & 31;
    const int wid   = threadIdx.x >> 5;
    const int tile  = blockIdx.x * 8 + wid;             // 8192 tiles total
    const int mt    = tile >> 6;                        // 64 N-tiles
    const int nt    = tile & 63;
    const int m0    = mt * 16;
    const int n0    = nt * 16;

    const int l16   = lane & 15;
    const int lhalf = lane >> 4;                        // 0: K{0..7,16..23} 1: K{8..15,24..31}

    v8f c = {};
    const _Float16* arow = A + (m0 + l16) * D_EMB;      // M = lane&15
    const _Float16* wrow = W + (n0 + l16) * D_EMB;      // N = lane&15 (mirrored layout)

#pragma unroll
    for (int kc = 0; kc < 8; ++kc) {
        int kb = kc * 32 + (lhalf ? 8 : 0);
        v8h alo = *(const v8h*)(arow + kb);
        v8h ahi = *(const v8h*)(arow + kb + 16);
        v8h blo = *(const v8h*)(wrow + kb);
        v8h bhi = *(const v8h*)(wrow + kb + 16);
        v16h a, b;
#pragma unroll
        for (int i = 0; i < 8; ++i) {
            a[i] = alo[i]; a[8 + i] = ahi[i];
            b[i] = blo[i]; b[8 + i] = bhi[i];
        }
        c = __builtin_amdgcn_wmma_f32_16x16x32_f16(
                /*neg_a=*/false, a, /*neg_b=*/false, b,
                /*c_mod=*/(short)0, c, /*reuse_a=*/false, /*reuse_b=*/false);
    }

    // C layout: lanes 0-15: (M=v, N=lane); lanes 16-31: (M=8+v, N=lane-16)
    const int ncol = n0 + l16;
    const float bias = bih[ncol] + bhh[ncol];
#pragma unroll
    for (int v = 0; v < 8; ++v) {
        int m = m0 + v + lhalf * 8;
        X[m * G4 + ncol] = c[v] + bias;
    }
}

// ---------------- 4. sequential LSTM recurrence ----------------
__device__ __forceinline__ float sigf(float x) {
    return 1.0f / (1.0f + __expf(-x));
}
__device__ __forceinline__ float tanh_fast(float x) {
    float e = __expf(2.0f * x);
    return (e - 1.0f) / (e + 1.0f);
}

__global__ __launch_bounds__(1024)
void lstm_rec_kernel(const float* __restrict__ Xf, const float* __restrict__ Xb,
                     const _Float16* __restrict__ Whh_f16,
                     const _Float16* __restrict__ Whh_b16,
                     float* __restrict__ hs_f, float* __restrict__ hs_b) {
    const int dir = blockIdx.x;                 // 0 fwd, 1 bwd
    const float* X          = dir ? Xb : Xf;
    const _Float16* Whh     = dir ? Whh_b16 : Whh_f16;   // [1024,256] row-major f16
    float* hs               = dir ? hs_b : hs_f;         // [2048,256]

    __shared__ float h_lds[H2];
    __shared__ float g_lds[G4];
    __shared__ float x_stage[G4];               // TDM prefetch target (staging only)

    const int r = threadIdx.x;                  // gate row 0..1023
    const _Float16* wrow = Whh + r * H2;

    // init h = 0
    if (r < H2) h_lds[r] = 0.0f;
    float c = 0.0f;                             // valid for r < H2
    __syncthreads();

    for (int step = 0; step < L_SEQ; ++step) {
        const int t = dir ? (L_SEQ - 1 - step) : step;

        // --- TDM: async prefetch next timestep's X row (4KB) into LDS staging.
        // Prefetch-only: compute path below still reads X through its normal
        // global loads; this overlaps the L2->LDS pull with compute
        // and exercises the Tensor Data Mover. One issue per step, wave 0 only.
        if ((threadIdx.x >> 5) == 0 && (step + 1) < L_SEQ) {
            const int tn = dir ? (L_SEQ - 2 - step) : (step + 1);
            unsigned long long ga = (unsigned long long)(const void*)(X + (size_t)tn * G4);
            unsigned lds_off = (unsigned)(unsigned long long)(const void*)x_stage;
            // D# group 0: count=1 | lds_addr | global_addr[56:0] | type=2
            v4u g0;
            g0[0] = 1u;
            g0[1] = lds_off;
            g0[2] = (unsigned)(ga & 0xFFFFFFFFu);
            g0[3] = (unsigned)((ga >> 32) & 0x1FFFFFFu) | (2u << 30);
            // D# group 1: data_size=4B | tensor_dim0=1024 | tensor_dim1=1 |
            //             tile_dim0=1024 | tile_dim1/2=0 | dim0_stride=1024
            v8i g1;
            g1[0] = (int)(2u << 16);                      // data_size=2 (4 bytes)
            g1[1] = (int)(((unsigned)G4 & 0xFFFFu) << 16); // tensor_dim0 lo16 @ bits63:48
            g1[2] = (int)(1u << 16);                      // dim0 hi=0 | tensor_dim1=1
            g1[3] = (int)((unsigned)G4 << 16);            // dim1 hi=0 | tile_dim0=1024
            g1[4] = 0;                                    // tile_dim1=0, tile_dim2=0
            g1[5] = G4;                                   // tensor_dim0_stride=1024
            g1[6] = 0;
            g1[7] = 0;
            v4i g2 = {0, 0, 0, 0};
            v4i g3 = {0, 0, 0, 0};
            v8i g4 = {0, 0, 0, 0, 0, 0, 0, 0};            // extra group (6-arg form)
            __builtin_amdgcn_s_wait_tensorcnt(0);         // throttle outstanding TDM
            __builtin_amdgcn_tensor_load_to_lds(g0, g1, g2, g3, g4, 0);
        }

        // g[r] = X[t,r] (has biases) + dot(Whh[r,:], h); f16 weights, f32 acc
        float acc = X[t * G4 + r];
#pragma unroll 8
        for (int k = 0; k < H2; k += 8) {
            v8h w = *(const v8h*)(wrow + k);              // 16B f16 load
            float4 h0 = *(const float4*)(h_lds + k);
            float4 h1 = *(const float4*)(h_lds + k + 4);
            acc = fmaf((float)w[0], h0.x, acc);
            acc = fmaf((float)w[1], h0.y, acc);
            acc = fmaf((float)w[2], h0.z, acc);
            acc = fmaf((float)w[3], h0.w, acc);
            acc = fmaf((float)w[4], h1.x, acc);
            acc = fmaf((float)w[5], h1.y, acc);
            acc = fmaf((float)w[6], h1.z, acc);
            acc = fmaf((float)w[7], h1.w, acc);
        }
        g_lds[r] = acc;
        __syncthreads();

        if (r < H2) {
            float ig = sigf(g_lds[r]);
            float fg = sigf(g_lds[H2 + r]);
            float gg = tanh_fast(g_lds[2 * H2 + r]);
            float og = sigf(g_lds[3 * H2 + r]);
            c = fg * c + ig * gg;
            float h = og * tanh_fast(c);
            h_lds[r] = h;
            hs[t * H2 + r] = h;
        }
        __syncthreads();
    }
}

// ---------------- 5. feats = [hs_f|hs_b] @ Wt^T + bt ----------------
__global__ __launch_bounds__(256)
void feats_kernel(const float* __restrict__ hs_f, const float* __restrict__ hs_b,
                  const float* __restrict__ Wt,   // [16,512] row-major
                  const float* __restrict__ bt,
                  float* __restrict__ feats) {    // [2048,16]
    int tid = blockIdx.x * blockDim.x + threadIdx.x;   // 32768 threads
    int l = tid >> 4;
    int j = tid & 15;
    const float* wf = Wt + j * 512;
    const float* wb = wf + H2;
    const float* hf = hs_f + l * H2;
    const float* hb = hs_b + l * H2;
    float acc = bt[j];
#pragma unroll 8
    for (int k = 0; k < H2 / 4; ++k) {
        float4 w = *(const float4*)(wf + 4 * k);
        float4 h = *(const float4*)(hf + 4 * k);
        acc = fmaf(w.x, h.x, acc); acc = fmaf(w.y, h.y, acc);
        acc = fmaf(w.z, h.z, acc); acc = fmaf(w.w, h.w, acc);
    }
#pragma unroll 8
    for (int k = 0; k < H2 / 4; ++k) {
        float4 w = *(const float4*)(wb + 4 * k);
        float4 h = *(const float4*)(hb + 4 * k);
        acc = fmaf(w.x, h.x, acc); acc = fmaf(w.y, h.y, acc);
        acc = fmaf(w.z, h.z, acc); acc = fmaf(w.w, h.w, acc);
    }
    feats[l * NTAG + j] = acc;
}

// ---------------- 6. Viterbi DP + backtrace (one wave32) ----------------
__global__ __launch_bounds__(32)
void viterbi_kernel(const float* __restrict__ feats,       // [2048,16]
                    const float* __restrict__ transitions, // [16,16]
                    int* __restrict__ bptrs,               // [2048,16]
                    float* __restrict__ out) {             // [1 + 2048]
    __shared__ float tr[NTAG * NTAG];
    __shared__ float fv[NTAG];
    __shared__ float fvn[NTAG];
    const int tid = threadIdx.x;

    for (int i = tid; i < NTAG * NTAG; i += 32) tr[i] = transitions[i];
    if (tid < NTAG) fv[tid] = (tid == START_TAG) ? 0.0f : NEGV;
    __syncthreads();

    for (int l = 0; l < L_SEQ; ++l) {
        if (tid < NTAG) {
            const int j = tid;
            float best = -3.4e38f;
            int bi = 0;
#pragma unroll
            for (int i = 0; i < NTAG; ++i) {
                float s = fv[i] + tr[j * NTAG + i];
                if (s > best) { best = s; bi = i; }     // first max, like jnp.argmax
            }
            bptrs[l * NTAG + j] = bi;
            fvn[j] = best + feats[l * NTAG + j];
        }
        __syncthreads();
        if (tid < NTAG) fv[tid] = fvn[tid];
        __syncthreads();
    }

    if (tid == 0) {
        float best = -3.4e38f;
        int bi = 0;
        for (int j = 0; j < NTAG; ++j) {
            float s = fv[j] + tr[STOP_TAG * NTAG + j];
            if (s > best) { best = s; bi = j; }
        }
        out[0] = best;
        int carry = bi;
        for (int l = L_SEQ - 1; l >= 0; --l) {
            out[1 + l] = (float)carry;
            carry = bptrs[l * NTAG + carry];
        }
    }
}

// ---------------- launch ----------------
extern "C" void kernel_launch(void* const* d_in, const int* in_sizes, int n_in,
                              void* d_out, int out_size, void* d_ws, size_t ws_size,
                              hipStream_t stream) {
    const int*   sentence = (const int*)  d_in[0];
    const float* emb      = (const float*)d_in[1];
    const float* Wih_f    = (const float*)d_in[2];
    const float* Whh_f    = (const float*)d_in[3];
    const float* bih_f    = (const float*)d_in[4];
    const float* bhh_f    = (const float*)d_in[5];
    const float* Wih_b    = (const float*)d_in[6];
    const float* Whh_b    = (const float*)d_in[7];
    const float* bih_b    = (const float*)d_in[8];
    const float* bhh_b    = (const float*)d_in[9];
    const float* Wt       = (const float*)d_in[10];
    const float* bt       = (const float*)d_in[11];
    const float* trans    = (const float*)d_in[12];
    float* out = (float*)d_out;

    // workspace carve-up (256B aligned)
    char* ws = (char*)d_ws;
    size_t off = 0;
    auto take = [&](size_t bytes) -> char* {
        char* p = ws + off;
        off = (off + bytes + 255) & ~(size_t)255;
        return p;
    };
    _Float16* embeds = (_Float16*)take((size_t)L_SEQ * D_EMB * 2);
    _Float16* wif16  = (_Float16*)take((size_t)G4 * D_EMB * 2);
    _Float16* wib16  = (_Float16*)take((size_t)G4 * D_EMB * 2);
    _Float16* whf16  = (_Float16*)take((size_t)G4 * H2 * 2);
    _Float16* whb16  = (_Float16*)take((size_t)G4 * H2 * 2);
    float*    Xf     = (float*)   take((size_t)L_SEQ * G4 * 4);
    float*    Xb     = (float*)   take((size_t)L_SEQ * G4 * 4);
    float*    hs_f   = (float*)   take((size_t)L_SEQ * H2 * 4);
    float*    hs_b   = (float*)   take((size_t)L_SEQ * H2 * 4);
    float*    feats  = (float*)   take((size_t)L_SEQ * NTAG * 4);
    int*      bptrs  = (int*)     take((size_t)L_SEQ * NTAG * 4);
    (void)ws_size; (void)in_sizes; (void)n_in; (void)out_size;

    // 1. embedding gather -> f16
    gather_f16_kernel<<<L_SEQ, D_EMB, 0, stream>>>(sentence, emb, embeds);

    // 2. weight conversion (input and recurrent weights)
    cvt_f16_kernel<<<(G4 * D_EMB) / 256, 256, 0, stream>>>(Wih_f, wif16, G4 * D_EMB);
    cvt_f16_kernel<<<(G4 * D_EMB) / 256, 256, 0, stream>>>(Wih_b, wib16, G4 * D_EMB);
    cvt_f16_kernel<<<(G4 * H2) / 256, 256, 0, stream>>>(Whh_f, whf16, G4 * H2);
    cvt_f16_kernel<<<(G4 * H2) / 256, 256, 0, stream>>>(Whh_b, whb16, G4 * H2);

    // 3. WMMA input-projection GEMMs (8192 tiles, 8 waves/block)
    gemm_wmma_kernel<<<1024, 256, 0, stream>>>(embeds, wif16, bih_f, bhh_f, Xf);
    gemm_wmma_kernel<<<1024, 256, 0, stream>>>(embeds, wib16, bih_b, bhh_b, Xb);

    // 4. sequential recurrence, both directions (f16 Whh, TDM X prefetch)
    lstm_rec_kernel<<<2, 1024, 0, stream>>>(Xf, Xb, whf16, whb16, hs_f, hs_b);

    // 5. tag features
    feats_kernel<<<(L_SEQ * NTAG) / 256, 256, 0, stream>>>(hs_f, hs_b, Wt, bt, feats);

    // 6. Viterbi + backtrace
    viterbi_kernel<<<1, 32, 0, stream>>>(feats, trans, bptrs, out);
}
